// ETEButterflyLayer_71433896067727
// MI455X (gfx1250) — compile-verified, matrix operand order checked by reading, not run on hardware
//
#include <hip/hip_runtime.h>

// ---------------------------------------------------------------------------
// ETEButterfly tree autoencoder on CDNA5 (gfx1250).
// All heavy math = bf16 WMMA (v_wmma_f32_16x16x32_bf16), f32 accumulate.
// Activations kept in bf16 in workspace to halve HBM traffic (mem-bound op).
// ---------------------------------------------------------------------------

typedef __attribute__((ext_vector_type(16))) __bf16 v16bf;
typedef __attribute__((ext_vector_type(8)))  __bf16 v8bf;
typedef __attribute__((ext_vector_type(8)))  float  v8f;

#define NLVL 8
#define C64  64
#define BATCH 256
#define OUT_ELEMS (256 * 8192)

static __device__ __forceinline__ v16bf join16(v8bf lo, v8bf hi) {
    v16bf r;
#pragma unroll
    for (int i = 0; i < 8; ++i) { r[i] = lo[i]; r[8 + i] = hi[i]; }
    return r;
}

static __device__ __forceinline__ v8f zero8() {
    v8f z;
#pragma unroll
    for (int i = 0; i < 8; ++i) z[i] = 0.0f;
    return z;
}

#define WMMA_BF16(a, b, c) \
    __builtin_amdgcn_wmma_f32_16x16x32_bf16(false, (a), false, (b), (short)0, (c), false, false)

// ---------------------------------------------------------------------------
// Encoder input conv: A=[65536,32] f32 (in_data view), W=[32,64] f32.
// One WMMA K-step (K=32).  Out: relu -> bf16, layout [B*256, 64].
// ---------------------------------------------------------------------------
__global__ __launch_bounds__(256) void k_enc_in(const float* __restrict__ x,
                                                const float* __restrict__ W,
                                                const float* __restrict__ bias,
                                                __bf16* __restrict__ dst) {
    __shared__ __bf16 Wt[64 * 32];   // transposed [d][r]
    __shared__ float  Bs[64];
    const int tid = threadIdx.x;
    for (int i = tid; i < 32 * 64; i += 256) {
        int r = i >> 6, d = i & 63;
        Wt[d * 32 + r] = (__bf16)W[i];
    }
    if (tid < 64) Bs[tid] = bias[tid];
    __syncthreads();

    const int wave = tid >> 5, lane = tid & 31;
    const int half = lane >> 4, lrow = lane & 15;
    const int m0 = blockIdx.x * 128 + wave * 16;
    const float* Ab = x + (size_t)(m0 + lrow) * 32;
    const int kbase = half * 8;

    v16bf a;
#pragma unroll
    for (int i = 0; i < 8; ++i) {
        a[i]     = (__bf16)Ab[kbase + i];
        a[8 + i] = (__bf16)Ab[kbase + 16 + i];
    }

    v8f acc[4];
#pragma unroll
    for (int n = 0; n < 4; ++n) acc[n] = zero8();

#pragma unroll
    for (int n = 0; n < 4; ++n) {
        const __bf16* Wp = &Wt[(n * 16 + lrow) * 32 + kbase];
        v8bf blo = *(const v8bf*)(Wp);
        v8bf bhi = *(const v8bf*)(Wp + 16);
        acc[n] = WMMA_BF16(a, join16(blo, bhi), acc[n]);
    }

#pragma unroll
    for (int n = 0; n < 4; ++n) {
        const int col = n * 16 + lrow;
        const float bv = Bs[col];
#pragma unroll
        for (int v = 0; v < 8; ++v) {
            const int mrow = m0 + v + half * 8;
            float xx = acc[n][v] + bv;
            xx = xx > 0.0f ? xx : 0.0f;
            dst[(size_t)mrow * 64 + col] = (__bf16)xx;
        }
    }
}

// ---------------------------------------------------------------------------
// One tree level.  Per branch k (blockIdx.y): contiguous A=[M,128] bf16 at
// parent (k>>1); W=[128,64] f32 -> LDS transposed bf16; out relu -> bf16 at
// dst + k*M*64.  grid = (M/128, K), 256 threads (8 waves x 16-row subtile).
// ---------------------------------------------------------------------------
__global__ __launch_bounds__(256) void k_tree(const __bf16* __restrict__ src,
                                              __bf16* __restrict__ dst,
                                              const float* __restrict__ Wl,   // level slice base
                                              const float* __restrict__ bl,   // level bias base
                                              int M) {
    __shared__ __bf16 Wt[64 * 128];  // [d][r]  16 KB
    __shared__ float  Bs[64];
    const int k   = blockIdx.y;
    const int tid = threadIdx.x;
    const float* Wg = Wl + (size_t)k * (2 * 64 * 64);
    for (int i = tid; i < 128 * 64; i += 256) {
        int r = i >> 6, d = i & 63;
        Wt[d * 128 + r] = (__bf16)Wg[i];
    }
    if (tid < 64) Bs[tid] = bl[(size_t)k * 64 + tid];
    __syncthreads();

    const int wave = tid >> 5, lane = tid & 31;
    const int half = lane >> 4, lrow = lane & 15;
    const int m0 = blockIdx.x * 128 + wave * 16;
    const __bf16* Abase = src + (size_t)(k >> 1) * ((size_t)M * 128)
                              + (size_t)(m0 + lrow) * 128;

    v8f acc[4];
#pragma unroll
    for (int n = 0; n < 4; ++n) acc[n] = zero8();

#pragma unroll
    for (int s = 0; s < 4; ++s) {
        const int kbase = s * 32 + half * 8;
        v8bf alo = *(const v8bf*)(Abase + kbase);
        v8bf ahi = *(const v8bf*)(Abase + kbase + 16);
        v16bf a = join16(alo, ahi);
#pragma unroll
        for (int n = 0; n < 4; ++n) {
            const __bf16* Wp = &Wt[(n * 16 + lrow) * 128 + kbase];
            v8bf blo = *(const v8bf*)(Wp);
            v8bf bhi = *(const v8bf*)(Wp + 16);
            acc[n] = WMMA_BF16(a, join16(blo, bhi), acc[n]);
        }
    }

    __bf16* Dbase = dst + (size_t)k * M * 64;
#pragma unroll
    for (int n = 0; n < 4; ++n) {
        const int col = n * 16 + lrow;
        const float bv = Bs[col];
#pragma unroll
        for (int v = 0; v < 8; ++v) {
            const int mrow = m0 + v + half * 8;
            float xx = acc[n][v] + bv;
            xx = xx > 0.0f ? xx : 0.0f;
            Dbase[(size_t)mrow * 64 + col] = (__bf16)xx;
        }
    }
}

// ---------------------------------------------------------------------------
// Mid projection: per branch k, [B,64] x [64,4] -> mid[b, k*4+m]  (f32).
// Tiny (33 MFLOP) -> VALU.  grid = 256 blocks (k), 256 threads (b).
// ---------------------------------------------------------------------------
__global__ __launch_bounds__(256) void k_mid(const __bf16* __restrict__ feat,
                                             const float* __restrict__ en_fea,
                                             float* __restrict__ mid) {
    __shared__ float Ws[64 * 4];
    const int k = blockIdx.x, b = threadIdx.x;
    if (b < 256) {
        if (b < 64 * 4) Ws[b] = en_fea[(size_t)k * 256 + b];
    }
    __syncthreads();
    const __bf16* f = feat + ((size_t)k * BATCH + b) * 64;
    float acc[4] = {0.f, 0.f, 0.f, 0.f};
#pragma unroll 8
    for (int c = 0; c < 64; ++c) {
        const float fv = (float)f[c];
#pragma unroll
        for (int m = 0; m < 4; ++m) acc[m] += fv * Ws[c * 4 + m];
    }
#pragma unroll
    for (int m = 0; m < 4; ++m)
        mid[(size_t)b * 1024 + k * 4 + m] = acc[m];
}

// ---------------------------------------------------------------------------
// Decoder input conv (K=2, folded mid_dense + transpose): row bb=i*256+b,
// position n:  a_f = mid[b, 4n+2f+i] * md[(2n+f)*2+i];
// out[(bb*256+n)*64+d] = relu(a0*W[0][d]+a1*W[1][d]+bias[d]) -> bf16.
// ---------------------------------------------------------------------------
__global__ __launch_bounds__(256) void k_dec_in(const float* __restrict__ mid,
                                                const float* __restrict__ md,   // [512][2]
                                                const float* __restrict__ W,    // [2][64]
                                                const float* __restrict__ bias, // [64]
                                                __bf16* __restrict__ dst) {
    __shared__ float Ws[128];
    __shared__ float Bs[64];
    const int tid = threadIdx.x;
    if (tid < 128) Ws[tid] = W[tid];
    if (tid < 64)  Bs[tid] = bias[tid];
    __syncthreads();

    const int g  = blockIdx.x * 256 + tid;   // 0 .. 131071
    const int bb = g >> 8;                   // 0 .. 511
    const int n  = g & 255;
    const int i  = bb >> 8;                  // real/imag
    const int b  = bb & 255;

    const float a0 = mid[(size_t)b * 1024 + 4 * n + i]     * md[(2 * n) * 2 + i];
    const float a1 = mid[(size_t)b * 1024 + 4 * n + 2 + i] * md[(2 * n + 1) * 2 + i];

    __bf16* o = dst + (size_t)g * 64;
#pragma unroll 8
    for (int d = 0; d < 64; ++d) {
        float xx = a0 * Ws[d] + a1 * Ws[64 + d] + Bs[d];
        xx = xx > 0.0f ? xx : 0.0f;
        o[d] = (__bf16)xx;
    }
}

// ---------------------------------------------------------------------------
// Out projection: per branch k, [512,64] bf16 x [64,64] -> scatter:
//   bb<256,  col even -> P[bb][k*32 + col/2]
//   bb>=256, col odd  -> Q[bb-256][k*32 + col/2]
// grid = (4, 256), 256 threads.  K loop: 2 steps of 32.
// ---------------------------------------------------------------------------
__global__ __launch_bounds__(256) void k_outproj(const __bf16* __restrict__ feat,
                                                 const float* __restrict__ de_fea,
                                                 float* __restrict__ P,
                                                 float* __restrict__ Q) {
    __shared__ __bf16 Wt[64 * 64];   // [o][c]  8 KB
    const int k   = blockIdx.y;
    const int tid = threadIdx.x;
    const float* Wg = de_fea + (size_t)k * 4096;   // [64 c][64 o]
    for (int i = tid; i < 64 * 64; i += 256) {
        int c = i >> 6, o = i & 63;
        Wt[o * 64 + c] = (__bf16)Wg[i];
    }
    __syncthreads();

    const int wave = tid >> 5, lane = tid & 31;
    const int half = lane >> 4, lrow = lane & 15;
    const int m0 = blockIdx.x * 128 + wave * 16;
    const __bf16* Abase = feat + (size_t)k * (512 * 64) + (size_t)(m0 + lrow) * 64;

    v8f acc[4];
#pragma unroll
    for (int n = 0; n < 4; ++n) acc[n] = zero8();

#pragma unroll
    for (int s = 0; s < 2; ++s) {
        const int kbase = s * 32 + half * 8;
        v8bf alo = *(const v8bf*)(Abase + kbase);
        v8bf ahi = *(const v8bf*)(Abase + kbase + 16);
        v16bf a = join16(alo, ahi);
#pragma unroll
        for (int n = 0; n < 4; ++n) {
            const __bf16* Wp = &Wt[(n * 16 + lrow) * 64 + kbase];
            v8bf blo = *(const v8bf*)(Wp);
            v8bf bhi = *(const v8bf*)(Wp + 16);
            acc[n] = WMMA_BF16(a, join16(blo, bhi), acc[n]);
        }
    }

#pragma unroll
    for (int n = 0; n < 4; ++n) {
        const int col = n * 16 + lrow;
        const int t   = col >> 1;
#pragma unroll
        for (int v = 0; v < 8; ++v) {
            const int bb = m0 + v + half * 8;
            const float val = acc[n][v];
            const size_t oidx = (size_t)(bb & 255) * 8192 + (size_t)k * 32 + t;
            if (bb < 256) {
                if ((col & 1) == 0) P[oidx] = val;
            } else {
                if ((col & 1) == 1) Q[oidx] = val;
            }
        }
    }
}

// ---------------------------------------------------------------------------
// Final: out = P - Q
// ---------------------------------------------------------------------------
__global__ __launch_bounds__(256) void k_combine(const float* __restrict__ P,
                                                 const float* __restrict__ Q,
                                                 float* __restrict__ out) {
    const int g = blockIdx.x * 256 + threadIdx.x;
    out[g] = P[g] - Q[g];
}

// ---------------------------------------------------------------------------
extern "C" void kernel_launch(void* const* d_in, const int* in_sizes, int n_in,
                              void* d_out, int out_size, void* d_ws, size_t ws_size,
                              hipStream_t stream) {
    (void)in_sizes; (void)n_in; (void)out_size; (void)ws_size;

    const float* in_data    = (const float*)d_in[0];
    const float* w_en_in    = (const float*)d_in[1];
    const float* b_en_in    = (const float*)d_in[2];
    const float* en_filters = (const float*)d_in[3];
    const float* en_biases  = (const float*)d_in[4];
    const float* en_fea     = (const float*)d_in[5];
    const float* mid_dense  = (const float*)d_in[6];
    const float* w_de_in    = (const float*)d_in[7];
    const float* b_de_in    = (const float*)d_in[8];
    const float* de_filters = (const float*)d_in[9];
    const float* de_biases  = (const float*)d_in[10];
    const float* de_fea     = (const float*)d_in[11];
    float* out = (float*)d_out;

    // workspace layout
    char* ws = (char*)d_ws;
    const size_t BUF_BYTES = 8388608ull * 2ull;           // 16 MiB each (bf16)
    __bf16* buf[2] = { (__bf16*)ws, (__bf16*)(ws + BUF_BYTES) };
    float*  mid    = (float*)(ws + 2 * BUF_BYTES);        // 1 MiB
    // P/Q overlay the free ping buffer after decoder level 8 (feat in buf[0])
    float* P = (float*)(ws + BUF_BYTES);
    float* Q = (float*)(ws + BUF_BYTES + 8388608ull);

    const size_t FILT_LVL = 256ull * 2 * 64 * 64;         // per-level filter slice
    const size_t BIAS_LVL = 256ull * 64;

    // 1. encoder input conv -> buf0  ([256][B*1][64] == [B*256,64])
    k_enc_in<<<512, 256, 0, stream>>>(in_data, w_en_in, b_en_in, buf[0]);

    // 2. encoder tree, levels 1..8  (B'=256)
    for (int lvl = 1; lvl <= NLVL; ++lvl) {
        const int L = 512 >> lvl;             // input length at this level
        const int K = 1 << lvl;               // branches
        const int M = 256 * (L >> 1);         // output rows per branch
        dim3 grid(M / 128, K);
        k_tree<<<grid, 256, 0, stream>>>(buf[(lvl - 1) & 1], buf[lvl & 1],
                                         en_filters + (size_t)(lvl - 1) * FILT_LVL,
                                         en_biases  + (size_t)(lvl - 1) * BIAS_LVL, M);
    }
    // feat_en in buf[0]  ([256][256][64])

    // 3. mid projection
    k_mid<<<256, 256, 0, stream>>>(buf[0], en_fea, mid);

    // 4. decoder input conv -> buf0  ([512*256, 64] bf16)
    k_dec_in<<<512, 256, 0, stream>>>(mid, mid_dense, w_de_in, b_de_in, buf[0]);

    // 5. decoder tree, levels 1..8  (B'=512)
    for (int lvl = 1; lvl <= NLVL; ++lvl) {
        const int L = 512 >> lvl;
        const int K = 1 << lvl;
        const int M = 512 * (L >> 1);
        dim3 grid(M / 128, K);
        k_tree<<<grid, 256, 0, stream>>>(buf[(lvl - 1) & 1], buf[lvl & 1],
                                         de_filters + (size_t)(lvl - 1) * FILT_LVL,
                                         de_biases  + (size_t)(lvl - 1) * BIAS_LVL, M);
    }
    // feat_de in buf[0]  ([256][512][64])

    // 6. out projection -> P/Q (in buf1 region)
    {
        dim3 grid(4, 256);
        k_outproj<<<grid, 256, 0, stream>>>(buf[0], de_fea, P, Q);
    }

    // 7. combine
    k_combine<<<OUT_ELEMS / 256, 256, 0, stream>>>(P, Q, out);
}
